// ImprovedListOpsMambaClassifier_14353780703720
// MI455X (gfx1250) — matrix operand order repaired and checked by reading, not compiled, not run on hardware
//
#include <hip/hip_runtime.h>
#include <hip/hip_fp16.h>
#include <math.h>

#define D_MODEL 256
#define D_INNER 512
#define D_STATE 16
#define D_CONVK 4
#define DT_RANK 16
#define N_LAYERS 4
#define BATCH 64
#define SEQL 2048

typedef __attribute__((ext_vector_type(16))) _Float16 v16h;
typedef __attribute__((ext_vector_type(8)))  float    v8f;

__device__ __forceinline__ float sigmoidf_(float x){ return 1.0f/(1.0f+__expf(-x)); }

// ---------------- embedding gather ----------------
__global__ void k_embed(const int* __restrict__ x, const float* __restrict__ emb,
                        float* __restrict__ h){
  size_t i = (size_t)blockIdx.x*blockDim.x + threadIdx.x;
  size_t total = (size_t)BATCH*SEQL*D_MODEL;
  if (i >= total) return;
  int c = (int)(i % D_MODEL);
  size_t r = i / D_MODEL;
  h[i] = emb[(size_t)x[r]*D_MODEL + c];
}

// ---------------- layernorm over D_MODEL=256, one wave per row ----------------
__global__ void k_layernorm(const float* __restrict__ in, float* __restrict__ out,
                            const float* __restrict__ w, const float* __restrict__ b,
                            int rows, size_t strideIn, size_t strideOut){
  int wave = threadIdx.x >> 5;
  int lane = threadIdx.x & 31;
  int row = blockIdx.x*8 + wave;
  if (row >= rows) return;
  const float* p = in + (size_t)row*strideIn;
  float v[8]; float s = 0.f, s2 = 0.f;
  #pragma unroll
  for (int j=0;j<8;++j){ float t = p[lane + j*32]; v[j]=t; s+=t; s2+=t*t; }
  #pragma unroll
  for (int m=16;m>=1;m>>=1){ s += __shfl_xor(s,m,32); s2 += __shfl_xor(s2,m,32); }
  float mean = s * (1.0f/D_MODEL);
  float var  = s2 * (1.0f/D_MODEL) - mean*mean;
  float rinv = rsqrtf(var + 1e-5f);
  float* q = out + (size_t)row*strideOut;
  #pragma unroll
  for (int j=0;j<8;++j){ int c = lane + j*32; q[c] = (v[j]-mean)*rinv*w[c] + b[c]; }
}

// ---------------- pre-swizzle weights into WMMA B-fragment layout ----------------
// B frag (16x16 tile, K-chunk of 32): lane holds col n=lane&15, elems e -> k=(e>>3)*16+(lane>>4)*8+(e&7)
__global__ void k_pack_b(const float* __restrict__ W, __half* __restrict__ Bp,
                         int K, int N, int ntTotal, int kcTotal){
  size_t i = (size_t)blockIdx.x*blockDim.x + threadIdx.x;
  size_t total = (size_t)kcTotal*ntTotal*32*16;
  if (i >= total) return;
  int e    = (int)(i & 15);
  int lane = (int)((i>>4) & 31);
  size_t r = i >> 9;
  int nt = (int)(r % ntTotal);
  int kc = (int)(r / ntTotal);
  int k = kc*32 + ((e>>3)<<4) + ((lane>>4)<<3) + (e&7);
  int n = nt*16 + (lane & 15);
  float val = (k < K && n < N) ? W[(size_t)k*N + n] : 0.0f;
  Bp[i] = __float2half(val);
}

// ---------------- WMMA GEMM: C[M,N] = A[M,K](f32) @ Bp(f16, pre-swizzled) ----------------
// 128 threads = 4 waves; wave owns 16 rows x (NT*16) cols; M must be a multiple of 64.
// Ping-pong double buffering of A (raw f32) and B fragments: loads for chunk kc+1 issue
// into the other register buffer while chunk kc's WMMAs execute (no WAR -> no v_nops,
// no s_wait_loadcnt 0 before the WMMAs). NTT = total 16-col tiles (compile-time) so B
// fragment addresses are base + immediate offsets.
// EPI: 0 none, 1 bias+softplus, 2 bias+relu, 3 bias.
template<int K, int NT, int NTT, int NACT, int EPI>
__global__ void k_gemm_wmma(const float* __restrict__ A, int lda,
                            const __half* __restrict__ Bp,
                            float* __restrict__ C, int ldc,
                            const float* __restrict__ bias){
  const int lane = threadIdx.x & 31;
  const int wave = threadIdx.x >> 5;
  const int rowBase = blockIdx.x*64 + wave*16;
  const int ntBase  = blockIdx.y*NT;
  const int row = rowBase + (lane & 15);
  const int kh  = lane >> 4;
  const float*  arow = A + (size_t)row*lda + kh*8;
  const __half* bptr = Bp + ((size_t)ntBase*32 + (size_t)lane)*16;

  v8f acc[NT];
  #pragma unroll
  for (int j=0;j<NT;++j)
    #pragma unroll
    for (int r=0;r<8;++r) acc[j][r] = 0.0f;

  constexpr int kcTotal = (K + 31) / 32;

  auto loadA = [&](int kc, float4* ra){
    const float* p = arow + kc*32;
    ra[0] = *reinterpret_cast<const float4*>(p);
    ra[1] = *reinterpret_cast<const float4*>(p + 4);
    ra[2] = *reinterpret_cast<const float4*>(p + 16);
    ra[3] = *reinterpret_cast<const float4*>(p + 20);
    __builtin_prefetch(p + 128, 0, 3);       // speculative, ~4 chunks ahead
  };
  auto loadB = [&](int kc, v16h* bf){
    const __half* p = bptr + (size_t)kc*(NTT*512);
    #pragma unroll
    for (int j=0;j<NT;++j)
      bf[j] = *reinterpret_cast<const v16h*>(p + j*512);   // immediate offsets
  };
  auto comp = [&](const float4* ra, const v16h* bf){
    v16h a;
    a[0]=(_Float16)ra[0].x; a[1]=(_Float16)ra[0].y; a[2]=(_Float16)ra[0].z; a[3]=(_Float16)ra[0].w;
    a[4]=(_Float16)ra[1].x; a[5]=(_Float16)ra[1].y; a[6]=(_Float16)ra[1].z; a[7]=(_Float16)ra[1].w;
    a[8]=(_Float16)ra[2].x; a[9]=(_Float16)ra[2].y; a[10]=(_Float16)ra[2].z; a[11]=(_Float16)ra[2].w;
    a[12]=(_Float16)ra[3].x; a[13]=(_Float16)ra[3].y; a[14]=(_Float16)ra[3].z; a[15]=(_Float16)ra[3].w;
    #pragma unroll
    for (int j=0;j<NT;++j)
      acc[j] = __builtin_amdgcn_wmma_f32_16x16x32_f16(
          false, a, false, bf[j], (short)0, acc[j], false, false);
  };

  if constexpr (K % 32 == 0){
    float4 ra0[4], ra1[4];
    v16h   b0[NT], b1[NT];
    loadA(0, ra0); loadB(0, b0);
    int kc = 0;
    for (; kc + 1 < kcTotal; kc += 2){
      loadA(kc+1, ra1); loadB(kc+1, b1);     // fill other buffer
      comp(ra0, b0);
      if (kc + 2 < kcTotal){ loadA(kc+2, ra0); loadB(kc+2, b0); }
      comp(ra1, b1);
    }
    if (kc < kcTotal) comp(ra0, b0);         // odd kcTotal tail
  } else {
    // K == DT_RANK (16): single K-chunk; upper half of the fragment is zero
    // (packed B is zero there too).
    float4 ra[2];
    ra[0] = *reinterpret_cast<const float4*>(arow);
    ra[1] = *reinterpret_cast<const float4*>(arow + 4);
    v16h a;
    a[0]=(_Float16)ra[0].x; a[1]=(_Float16)ra[0].y; a[2]=(_Float16)ra[0].z; a[3]=(_Float16)ra[0].w;
    a[4]=(_Float16)ra[1].x; a[5]=(_Float16)ra[1].y; a[6]=(_Float16)ra[1].z; a[7]=(_Float16)ra[1].w;
    #pragma unroll
    for (int e=8;e<16;++e) a[e] = (_Float16)0.0f;
    v16h b0[NT];
    loadB(0, b0);
    #pragma unroll
    for (int j=0;j<NT;++j)
      acc[j] = __builtin_amdgcn_wmma_f32_16x16x32_f16(
          false, a, false, b0[j], (short)0, acc[j], false, false);
  }

  #pragma unroll
  for (int j=0;j<NT;++j){
    const int n = (ntBase + j)*16 + (lane & 15);
    if constexpr (NACT % 16 != 0){ if (n >= NACT) continue; }
    float bv = (EPI != 0) ? bias[n] : 0.0f;
    #pragma unroll
    for (int r=0;r<8;++r){
      const int m = rowBase + kh*8 + r;  // C/D layout: lanes 0-15 -> M=r, 16-31 -> M=r+8
      float v = acc[j][r] + bv;
      if (EPI == 1)      v = (v > 20.0f) ? v : logf(1.0f + __expf(v));
      else if (EPI == 2) v = fmaxf(v, 0.0f);
      C[(size_t)m*ldc + n] = v;
    }
  }
}

// ---------------- depthwise causal conv (k=4) + SiLU, reads xin half of xz ----------------
__global__ void k_conv_silu(const float* __restrict__ xz, const float* __restrict__ cw,
                            const float* __restrict__ cb, float* __restrict__ xc){
  size_t i = (size_t)blockIdx.x*blockDim.x + threadIdx.x;
  size_t total = (size_t)BATCH*SEQL*D_INNER;
  if (i >= total) return;
  int d = (int)(i % D_INNER);
  size_t ml = i / D_INNER;
  int l = (int)(ml % SEQL);
  size_t b = ml / SEQL;
  float acc = cb[d];
  #pragma unroll
  for (int j=0;j<D_CONVK;++j){
    int lp = l - (D_CONVK-1) + j;
    if (lp >= 0)
      acc += cw[d*D_CONVK + j] * xz[((size_t)b*SEQL + lp)*(2*D_INNER) + d];
  }
  xc[i] = acc * sigmoidf_(acc);
}

// ---------------- selective scan: register state; per-timestep B/C row pipelined ----------
// through LDS with async global->LDS copies, double-buffered one step ahead.
__global__ void k_scan(const float* __restrict__ delta, const float* __restrict__ xdbl,
                       const float* __restrict__ xc, const float* __restrict__ xz,
                       const float* __restrict__ A_log, const float* __restrict__ Dp,
                       float* __restrict__ y){
  __shared__ float sbc[8][2][32];                 // [wave][buf][B(16)++C(16)]
  const int d = blockIdx.x*blockDim.x + threadIdx.x;   // 0..511
  const int b = blockIdx.y;
  const int lane = threadIdx.x & 31;
  const int wv = threadIdx.x >> 5;

  float a[D_STATE];
  #pragma unroll
  for (int s=0;s<D_STATE;++s) a[s] = -__expf(A_log[d*D_STATE + s]);
  const float Dd = Dp[d];

  float h[D_STATE];
  #pragma unroll
  for (int s=0;s<D_STATE;++s) h[s] = 0.0f;

  const size_t rowBase = (size_t)b*SEQL;

  // async copy of x_dbl[row, 16+lane] (B++C) into this wave's LDS buffer (t&1)
  auto issue = [&](int t){
    unsigned loff = (unsigned)(size_t)&sbc[wv][t & 1][lane];  // flat low 32b == LDS offset
    unsigned long long ga =
        (unsigned long long)(xdbl + (rowBase + (size_t)t)*48 + DT_RANK + lane);
    asm volatile("global_load_async_to_lds_b32 %0, %1, off"
                 :: "v"(loff), "v"(ga) : "memory");
  };

  issue(0);
  for (int t=0; t<SEQL; ++t){
    const size_t r = rowBase + t;
    const float dt = delta[r*D_INNER + d];
    const float u  = xc[r*D_INNER + d];
    const float z  = xz[r*(2*D_INNER) + D_INNER + d];

    if (t + 1 < SEQL){
      issue(t + 1);                                          // keep next row in flight
      asm volatile("s_wait_asynccnt 0x1" ::: "memory");      // in-order: current buf ready
    } else {
      asm volatile("s_wait_asynccnt 0x0" ::: "memory");
    }
    const float* bcp = &sbc[wv][t & 1][0];
    float bcv[32];
    #pragma unroll
    for (int q=0;q<8;++q)
      *reinterpret_cast<float4*>(bcv + q*4) = *reinterpret_cast<const float4*>(bcp + q*4);

    const float du = dt*u;
    float yv = 0.0f;
    #pragma unroll
    for (int s=0;s<D_STATE;++s){
      float hs = h[s]*__expf(dt*a[s]) + du*bcv[s];
      h[s] = hs;
      yv += hs*bcv[D_STATE + s];
    }
    yv = (yv + u*Dd) * (z * sigmoidf_(z));   // + xin*D, gate by silu(z), fused
    y[r*D_INNER + d] = yv;
  }
}

// ---------------- residual add + running-mean blend (cumsum along L), writes new h ----------
__global__ void k_resid_cumsum(const float* __restrict__ outb, float* __restrict__ h){
  const int c = threadIdx.x;        // 256
  const int b = blockIdx.x;         // 64
  float cs = 0.0f;
  for (int l=0;l<SEQL;++l){
    size_t idx = ((size_t)b*SEQL + l)*D_MODEL + c;
    float v = outb[idx] + h[idx];
    cs += v;
    h[idx] = 0.9f*v + 0.1f*cs/(float)(l+1);
  }
}

// =====================================================================================
extern "C" void kernel_launch(void* const* d_in, const int* in_sizes, int n_in,
                              void* d_out, int out_size, void* d_ws, size_t ws_size,
                              hipStream_t stream){
  (void)in_sizes; (void)n_in; (void)out_size; (void)ws_size;

  const int*   x      = (const int*)  d_in[0];
  const float* emb    = (const float*)d_in[1];
  const float* ln_w   = (const float*)d_in[2];
  const float* ln_b   = (const float*)d_in[3];
  const float* W_in   = (const float*)d_in[4];
  const float* conv_w = (const float*)d_in[5];
  const float* conv_b = (const float*)d_in[6];
  const float* W_x    = (const float*)d_in[7];
  const float* W_dt   = (const float*)d_in[8];
  const float* b_dt   = (const float*)d_in[9];
  const float* A_log  = (const float*)d_in[10];
  const float* Dpar   = (const float*)d_in[11];
  const float* W_out  = (const float*)d_in[12];
  const float* norm_w = (const float*)d_in[13];
  const float* norm_b = (const float*)d_in[14];
  const float* Wc1    = (const float*)d_in[15];
  const float* bc1    = (const float*)d_in[16];
  const float* Wc2    = (const float*)d_in[17];
  const float* bc2    = (const float*)d_in[18];

  char* ws = (char*)d_ws;
  size_t off = 0;
  auto alloc = [&](size_t bytes)->char*{
    off = (off + 255) & ~(size_t)255;
    char* p = ws + off; off += bytes; return p;
  };
  const size_t M = (size_t)BATCH*SEQL;            // 131072
  auto pk = [](int K,int N)->size_t{ return (size_t)((K+31)/32)*((N+15)/16)*512; };

  float* bh    = (float*)alloc(M*D_MODEL*4);      // hidden / residual
  float* bln   = (float*)alloc(M*D_MODEL*4);      // LN out; reused as W_out GEMM out
  float* bxz   = (float*)alloc(M*2*D_INNER*4);    // [xin | z]
  float* bxc   = (float*)alloc(M*D_INNER*4);      // conv+silu out
  float* bxdbl = (float*)alloc(M*48*4);           // [dt | B | C]
  float* bdel  = (float*)alloc(M*D_INNER*4);      // softplus(delta)
  float* by    = (float*)alloc(M*D_INNER*4);      // gated scan output
  float* blast = (float*)alloc(64*256*4);
  float* bhid  = (float*)alloc(64*128*4);
  __half* pWin = (__half*)alloc(N_LAYERS*pk(256,1024)*2);
  __half* pWx  = (__half*)alloc(N_LAYERS*pk(512,48)*2);
  __half* pWdt = (__half*)alloc(N_LAYERS*pk(16,512)*2);
  __half* pWo  = (__half*)alloc(N_LAYERS*pk(512,256)*2);
  __half* pC1  = (__half*)alloc(pk(256,128)*2);
  __half* pC2  = (__half*)alloc(pk(128,10)*2);

  auto packLaunch = [&](const float* W, __half* P, int K, int N){
    size_t elems = pk(K,N);
    k_pack_b<<<dim3((unsigned)((elems+255)/256)), dim3(256), 0, stream>>>(
        W, P, K, N, (N+15)/16, (K+31)/32);
  };
  for (int i=0;i<N_LAYERS;++i){
    packLaunch(W_in  + (size_t)i*256*1024, pWin + (size_t)i*pk(256,1024), 256, 1024);
    packLaunch(W_x   + (size_t)i*512*48,   pWx  + (size_t)i*pk(512,48),   512, 48);
    packLaunch(W_dt  + (size_t)i*16*512,   pWdt + (size_t)i*pk(16,512),   16,  512);
    packLaunch(W_out + (size_t)i*512*256,  pWo  + (size_t)i*pk(512,256),  512, 256);
  }
  packLaunch(Wc1, pC1, 256, 128);
  packLaunch(Wc2, pC2, 128, 10);

  k_embed<<<dim3((unsigned)((M*D_MODEL+255)/256)), 256, 0, stream>>>(x, emb, bh);

  for (int i=0;i<N_LAYERS;++i){
    k_layernorm<<<dim3((unsigned)(M/8)), 256, 0, stream>>>(
        bh, bln, ln_w + (size_t)i*256, ln_b + (size_t)i*256,
        (int)M, (size_t)256, (size_t)256);
    // h_ln @ W_in -> xz  (K=256, N=1024, 64 tiles)
    k_gemm_wmma<256,4,64,1024,0><<<dim3((unsigned)(M/64), 16), 128, 0, stream>>>(
        bln, 256, pWin + (size_t)i*pk(256,1024), bxz, 1024, nullptr);
    k_conv_silu<<<dim3((unsigned)((M*D_INNER+255)/256)), 256, 0, stream>>>(
        bxz, conv_w + (size_t)i*D_INNER*D_CONVK, conv_b + (size_t)i*D_INNER, bxc);
    // xin @ W_x -> x_dbl  (K=512, N=48, 3 tiles)
    k_gemm_wmma<512,3,3,48,0><<<dim3((unsigned)(M/64), 1), 128, 0, stream>>>(
        bxc, 512, pWx + (size_t)i*pk(512,48), bxdbl, 48, nullptr);
    // dt @ W_dt + b_dt -> softplus -> delta  (K=16 padded, N=512, 32 tiles)
    k_gemm_wmma<16,4,32,512,1><<<dim3((unsigned)(M/64), 8), 128, 0, stream>>>(
        bxdbl, 48, pWdt + (size_t)i*pk(16,512), bdel, 512, b_dt + (size_t)i*512);
    k_scan<<<dim3(2, 64), 256, 0, stream>>>(
        bdel, bxdbl, bxc, bxz, A_log + (size_t)i*D_INNER*D_STATE,
        Dpar + (size_t)i*D_INNER, by);
    // y @ W_out -> out (K=512, N=256, 16 tiles); reuse bln as out buffer
    k_gemm_wmma<512,4,16,256,0><<<dim3((unsigned)(M/64), 4), 128, 0, stream>>>(
        by, 512, pWo + (size_t)i*pk(512,256), bln, 256, nullptr);
    k_resid_cumsum<<<dim3(64), 256, 0, stream>>>(bln, bh);
  }

  // last token -> LN -> classifier
  k_layernorm<<<dim3(8), 256, 0, stream>>>(
      bh + (size_t)(SEQL-1)*D_MODEL, blast, norm_w, norm_b,
      64, (size_t)SEQL*D_MODEL, (size_t)256);
  k_gemm_wmma<256,4,8,128,2><<<dim3(1, 2), 128, 0, stream>>>(
      blast, 256, pC1, bhid, 128, bc1);
  k_gemm_wmma<128,1,1,10,3><<<dim3(1, 1), 128, 0, stream>>>(
      bhid, 128, pC2, (float*)d_out, 10, bc2);
}
